// LAINRDecoderSCENT_88441966559244
// MI455X (gfx1250) — matrix operand order repaired and checked by reading, not compiled, not run on hardware
//
#include <hip/hip_runtime.h>
#include <hip/hip_bf16.h>

typedef __attribute__((ext_vector_type(16))) _Float16 v16h;
typedef __attribute__((ext_vector_type(8)))  _Float16 h8;
typedef __attribute__((ext_vector_type(2)))  _Float16 h2;
typedef __attribute__((ext_vector_type(8)))  float    v8f;

// ---------------------------------------------------------------------------
// Generic batched WMMA GEMM:  C[z] = act( scale*(A[z] @ B[z]) + bias + biasMat + residual[z] )
// A: (M x K) f32 row-major (lda), B: (K x N) row-major (ldb) or (N x K) if transB.
// REQUIRES: M % 64 == 0, N % 64 == 0, K % 32 == 0, lda/ldb/ldc % 4 == 0.
// z -> b = z/nH, h = z%nH; per-operand offsets  b*s?b + h*s?h  (elements).
// ---------------------------------------------------------------------------
struct GemmArgs {
  const float* A; const float* B; float* C;
  const float* bias; const float* biasMat; const float* residual;
  long sAb, sAh, sBb, sBh, sCb, sCh;
  int lda, ldb, ldc, ldbm;
  int M, N, K;
  int transB, nH, act;   // act: 0 none, 1 relu
  float scale;
};

#define TM 64
#define TN 64
#define TK 32
#define LDSP 40  // padded f16 row stride (80B: 16B-aligned, bank-skewed)

__device__ __forceinline__ h8 cvt8(float4 a, float4 b) {
  h8 r;
  r[0] = (_Float16)a.x; r[1] = (_Float16)a.y; r[2] = (_Float16)a.z; r[3] = (_Float16)a.w;
  r[4] = (_Float16)b.x; r[5] = (_Float16)b.y; r[6] = (_Float16)b.z; r[7] = (_Float16)b.w;
  return r;
}

__global__ __launch_bounds__(128) void gemm_wmma_kernel(GemmArgs g) {
  __shared__ _Float16 As[TM][LDSP];
  __shared__ _Float16 Bs[TN][LDSP];   // Bs[n][k] (B stored N-major in LDS)

  const int z  = blockIdx.z;
  const int bb = z / g.nH, hh = z % g.nH;
  const float* A = g.A + (long)bb * g.sAb + (long)hh * g.sAh;
  const float* B = g.B + (long)bb * g.sBb + (long)hh * g.sBh;
  const long coff = (long)bb * g.sCb + (long)hh * g.sCh;
  float* C = g.C + coff;
  const float* R = g.residual ? (g.residual + coff) : nullptr;

  const int m0 = blockIdx.y * TM;
  const int n0 = blockIdx.x * TN;
  const int tid  = threadIdx.x;
  const int lane = tid & 31;
  const int wave = tid >> 5;
  const int wm = wave >> 1, wn = wave & 1;

  // staging coordinates ------------------------------------------------------
  const int ar = tid >> 1, ah = tid & 1;              // A: row 0..63, k-half 0/1
  const float* Aptr = A + (long)(m0 + ar) * g.lda + ah * 16;

  const int bn  = tid >> 1, bhf = tid & 1;            // B transposed source (N x K)
  const int bkp = tid >> 3, bnq = (tid & 7) * 8;      // B row-major source (K x N)
  const float* BptrT = B + (long)(n0 + bn) * g.ldb + bhf * 16;
  const float* BptrN = B + (long)(2 * bkp) * g.ldb + n0 + bnq;

  float4 a0, a1, a2, a3, b0, b1, b2, b3;

  auto loadTile = [&](int k0) {
    const float4* pa = (const float4*)(Aptr + k0);
    a0 = pa[0]; a1 = pa[1]; a2 = pa[2]; a3 = pa[3];
    if (g.transB) {
      const float4* pb = (const float4*)(BptrT + k0);
      b0 = pb[0]; b1 = pb[1]; b2 = pb[2]; b3 = pb[3];
    } else {
      const float4* p0 = (const float4*)(BptrN + (long)k0 * g.ldb);
      const float4* p1 = (const float4*)(BptrN + (long)(k0 + 1) * g.ldb);
      b0 = p0[0]; b1 = p0[1]; b2 = p1[0]; b3 = p1[1];
    }
  };

  auto storeTile = [&]() {
    h8* pa = (h8*)&As[ar][ah * 16];
    pa[0] = cvt8(a0, a1); pa[1] = cvt8(a2, a3);
    if (g.transB) {
      h8* pb = (h8*)&Bs[bn][bhf * 16];
      pb[0] = cvt8(b0, b1); pb[1] = cvt8(b2, b3);
    } else {
      // rows k=2*bkp (b0,b1) and k=2*bkp+1 (b2,b3), n = n0+bnq .. +8
      h8 e = cvt8(b0, b1), o = cvt8(b2, b3);
      #pragma unroll
      for (int j = 0; j < 8; ++j) {
        h2 v; v[0] = e[j]; v[1] = o[j];
        *(h2*)&Bs[bnq + j][2 * bkp] = v;
      }
    }
  };

  const v8f vzero = {0.f,0.f,0.f,0.f,0.f,0.f,0.f,0.f};
  v8f acc[2][2];
  acc[0][0] = vzero; acc[0][1] = vzero; acc[1][0] = vzero; acc[1][1] = vzero;

  const int rlo = lane & 15;
  const int hi  = lane >> 4;

  loadTile(0);
  for (int k0 = 0; k0 < g.K; k0 += TK) {
    storeTile();
    __syncthreads();
    const int kn = k0 + TK;
    if (kn < g.K) loadTile(kn);   // next tile in flight during WMMA

    v16h af[2], bf[2];
    #pragma unroll
    for (int t = 0; t < 2; ++t) {
      const int row = wm * 32 + t * 16 + rlo;
      #pragma unroll
      for (int e = 0; e < 8; ++e) {
        af[t][e]     = As[row][hi * 16 + e];
        af[t][e + 8] = As[row][hi * 16 + 8 + e];
      }
      const int col = wn * 32 + t * 16 + rlo;
      #pragma unroll
      for (int e = 0; e < 16; ++e)
        bf[t][e] = Bs[col][hi * 16 + e];
    }
    #pragma unroll
    for (int ri = 0; ri < 2; ++ri)
      #pragma unroll
      for (int ci = 0; ci < 2; ++ci)
        acc[ri][ci] = __builtin_amdgcn_wmma_f32_16x16x32_f16(
            false, af[ri], false, bf[ci], (short)0, acc[ri][ci], false, false);
    __syncthreads();
  }

  // epilogue (C/D layout: M = v + 8*hi, N = lane&15); dims are exact multiples
  #pragma unroll
  for (int ri = 0; ri < 2; ++ri)
    #pragma unroll
    for (int ci = 0; ci < 2; ++ci) {
      const int col = n0 + wn * 32 + ci * 16 + rlo;
      #pragma unroll
      for (int v = 0; v < 8; ++v) {
        const int row = m0 + wm * 32 + ri * 16 + hi * 8 + v;
        float val = acc[ri][ci][v] * g.scale;
        if (g.bias)    val += g.bias[col];
        if (g.biasMat) val += g.biasMat[(long)row * g.ldbm + col];
        if (R)         val += R[(long)row * g.ldc + col];
        if (g.act == 1) val = fmaxf(val, 0.f);
        C[(long)row * g.ldc + col] = val;
      }
    }
}

// ---------------------------------------------------------------------------
// LayerNorm over last dim W, one block (256 thr) per row.
// ---------------------------------------------------------------------------
__global__ __launch_bounds__(256) void ln_kernel(const float* __restrict__ x,
                                                 const float* __restrict__ g,
                                                 const float* __restrict__ b,
                                                 float* __restrict__ y, int W) {
  __shared__ float sm[256];
  const long row = blockIdx.x;
  const float* xr = x + row * W;
  float* yr = y + row * W;
  float s = 0.f;
  for (int c = threadIdx.x; c < W; c += 256) s += xr[c];
  sm[threadIdx.x] = s; __syncthreads();
  for (int st = 128; st > 0; st >>= 1) { if (threadIdx.x < st) sm[threadIdx.x] += sm[threadIdx.x + st]; __syncthreads(); }
  const float mu = sm[0] / (float)W;
  __syncthreads();
  float s2 = 0.f;
  for (int c = threadIdx.x; c < W; c += 256) { float d = xr[c] - mu; s2 += d * d; }
  sm[threadIdx.x] = s2; __syncthreads();
  for (int st = 128; st > 0; st >>= 1) { if (threadIdx.x < st) sm[threadIdx.x] += sm[threadIdx.x + st]; __syncthreads(); }
  const float inv = rsqrtf(sm[0] / (float)W + 1e-5f);
  for (int c = threadIdx.x; c < W; c += 256) yr[c] = (xr[c] - mu) * inv * g[c] + b[c];
}

// ---------------------------------------------------------------------------
// Row softmax over width W, in place; one block per row.
// ---------------------------------------------------------------------------
__global__ __launch_bounds__(256) void softmax_kernel(float* __restrict__ x, int W) {
  __shared__ float sm[256];
  const long row = blockIdx.x;
  float* xr = x + row * W;
  float m = -3.4e38f;
  for (int c = threadIdx.x; c < W; c += 256) m = fmaxf(m, xr[c]);
  sm[threadIdx.x] = m; __syncthreads();
  for (int st = 128; st > 0; st >>= 1) { if (threadIdx.x < st) sm[threadIdx.x] = fmaxf(sm[threadIdx.x], sm[threadIdx.x + st]); __syncthreads(); }
  const float mx = sm[0];
  __syncthreads();
  float s = 0.f;
  for (int c = threadIdx.x; c < W; c += 256) { float e = expf(xr[c] - mx); xr[c] = e; s += e; }
  sm[threadIdx.x] = s; __syncthreads();
  for (int st = 128; st > 0; st >>= 1) { if (threadIdx.x < st) sm[threadIdx.x] += sm[threadIdx.x + st]; __syncthreads(); }
  const float inv = 1.f / sm[0];
  for (int c = threadIdx.x; c < W; c += 256) xr[c] *= inv;
}

// ---------------------------------------------------------------------------
// Elementwise helpers
// ---------------------------------------------------------------------------
__global__ void add_kernel(const float* a, const float* b, float* c, long n, int relu) {
  long i = (long)blockIdx.x * blockDim.x + threadIdx.x;
  if (i < n) { float v = a[i] + b[i]; if (relu) v = fmaxf(v, 0.f); c[i] = v; }
}

// GEGLU: G[r][c] = H[r][c] * gelu(H[r][2048+c]), exact erf GELU; H is 2048x4096.
__global__ void geglu_kernel(const float* __restrict__ H, float* __restrict__ G, long n) {
  long i = (long)blockIdx.x * blockDim.x + threadIdx.x;
  if (i >= n) return;
  long r = i >> 11; int c = (int)(i & 2047);
  float a = H[r * 4096 + c];
  float x = H[r * 4096 + 2048 + c];
  float gl = 0.5f * x * (1.f + erff(x * 0.70710678118654752f));
  G[r * 2048 + c] = a * gl;
}

// Fourier features: coords (1024,2), Bm (32,2) -> out (1024,64) = [cos|sin](2*pi*coords@Bm^T)
__global__ void fourier_kernel(const float* __restrict__ coords, const float* __restrict__ Bm,
                               float* __restrict__ out) {
  int j = blockIdx.x * blockDim.x + threadIdx.x;
  if (j >= 1024) return;
  float x = coords[j * 2], y = coords[j * 2 + 1];
  for (int f = 0; f < 32; ++f) {
    float p = 6.283185307179586f * (x * Bm[f * 2] + y * Bm[f * 2 + 1]);
    out[j * 64 + f]      = cosf(p);
    out[j * 64 + 32 + f] = sinf(p);
  }
}

// Spatial bias: rel[i][j] = -ALPHA * (t[j] - pos[i])^2  (untransposed add, per reference)
__global__ void rel_kernel(const float* __restrict__ coords, float* __restrict__ rel) {
  long idx = (long)blockIdx.x * blockDim.x + threadIdx.x;
  if (idx >= 1024L * 1024L) return;
  int i = (int)(idx >> 10), j = (int)(idx & 1023);
  float gx = coords[j * 2], gy = coords[j * 2 + 1];
  int r = (int)(gx * 32.f); r = min(max(r, 0), 31);
  int c = (int)(gy * 32.f); c = min(max(c, 0), 31);
  float t   = (float)(r * 32 + c) * (1.f / 1024.f);
  float pos = ((float)i + 0.5f) * (1.f / 1024.f);
  float d = t - pos;
  rel[idx] = -10.f * d * d;
}

// Output projection: out[row,0..2] (+)= X[row,:]@W(512x3) + b
__global__ void outproj_kernel(const float* __restrict__ X, const float* __restrict__ W,
                               const float* __restrict__ b, float* __restrict__ out, int accum) {
  int idx = blockIdx.x * blockDim.x + threadIdx.x;
  if (idx >= 2048 * 3) return;
  int row = idx / 3, o = idx % 3;
  const float* xr = X + (long)row * 512;
  float s = b[o];
  for (int k = 0; k < 512; ++k) s += xr[k] * W[k * 3 + o];
  if (accum) out[idx] += s; else out[idx] = s;
}

// ---------------------------------------------------------------------------
// Host orchestration
// ---------------------------------------------------------------------------
static inline void launch_gemm(hipStream_t st,
    const float* A, const float* B, float* C,
    int M, int N, int K, int lda, int ldb, int ldc,
    int transB, float scale, const float* bias, int act,
    const float* residual, const float* biasMat, int ldbm,
    int Z, int nH,
    long sAb, long sAh, long sBb, long sBh, long sCb, long sCh) {
  GemmArgs g;
  g.A = A; g.B = B; g.C = C; g.bias = bias; g.biasMat = biasMat; g.residual = residual;
  g.sAb = sAb; g.sAh = sAh; g.sBb = sBb; g.sBh = sBh; g.sCb = sCb; g.sCh = sCh;
  g.lda = lda; g.ldb = ldb; g.ldc = ldc; g.ldbm = ldbm;
  g.M = M; g.N = N; g.K = K; g.transB = transB; g.nH = nH; g.act = act; g.scale = scale;
  dim3 grid((N + TN - 1) / TN, (M + TM - 1) / TM, Z);
  gemm_wmma_kernel<<<grid, dim3(128), 0, st>>>(g);
}

extern "C" void kernel_launch(void* const* d_in, const int* in_sizes, int n_in,
                              void* d_out, int out_size, void* d_ws, size_t ws_size,
                              hipStream_t stream) {
  (void)in_sizes; (void)n_in; (void)out_size; (void)ws_size;
  const float* coords = (const float*)d_in[0];   // (2,32,32,2); grid = first 1024 pairs
  const float* tokens = (const float*)d_in[1];   // (2,1024,256)
  int p = 2;
  auto nx = [&]() { return (const float*)d_in[p++]; };

  const float* B_q  = nx();
  const float* B_l0 = nx();
  const float* B_l1 = nx();
  const float* ql_w = nx(); const float* ql_b = nx();
  const float *ca_ng = nx(), *ca_nb = nx(), *ca_cg = nx(), *ca_cb = nx();
  const float *ca_q = nx(), *ca_kv = nx(), *ca_ow = nx(), *ca_ob = nx();

  struct Blk { const float *ng,*nb,*tq,*tkv,*tow,*tob,*fng,*fnb,*w1,*b1,*w2,*b2; };
  auto readBlk = [&]() {
    Blk b;
    b.ng = nx(); b.nb = nx(); b.tq = nx(); b.tkv = nx(); b.tow = nx(); b.tob = nx();
    b.fng = nx(); b.fnb = nx(); b.w1 = nx(); b.b1 = nx(); b.w2 = nx(); b.b2 = nx();
    return b;
  };
  struct Seq { const float *lw, *lb; Blk blk[2]; };
  auto readSeq = [&]() { Seq s; s.lw = nx(); s.lb = nx(); s.blk[0] = readBlk(); s.blk[1] = readBlk(); return s; };

  Seq bw0 = readSeq(), bw1 = readSeq();
  Seq md0 = readSeq(), md1 = readSeq();
  Blk hv = readBlk();
  const float* fsw0 = nx(); const float* fsb0 = nx();
  const float* fsw1 = nx(); const float* fsb1 = nx();
  const float* ow0 = nx(); const float* ob0 = nx();
  const float* ow1 = nx(); const float* ob1 = nx();

  float* out = (float*)d_out;   // (2,32,32,3) = 6144

  // workspace layout (floats)
  float* w = (float*)d_ws;
  size_t off = 0;
  auto alloc = [&](size_t n) { float* r = w + off; off += n; return r; };
  float* FENCQ = alloc(1024 * 64);
  float* FENC0 = alloc(1024 * 64);
  float* FENC1 = alloc(1024 * 64);
  float* REL   = alloc(1024L * 1024);
  float* XQ    = alloc(2048L * 512);
  float* TL    = alloc(2048L * 512);
  float* TT    = alloc(2048L * 256);
  float* QB    = alloc(2048L * 512);
  float* KVB   = alloc(2048L * 1024);
  float* OB    = alloc(2048L * 512);
  float* MODV  = alloc(2048L * 512);
  float* XB    = alloc(2048L * 512);
  float* XM    = alloc(2048L * 512);
  float* MODS0 = alloc(2048L * 512);
  float* MODS1 = alloc(2048L * 512);
  float* XV    = alloc(2048L * 512);
  float* SKIP  = alloc(2048L * 512);
  float* SIMR  = alloc(12L * 1024 * 1024);   // sim (8 heads) or H(8M)+G(4M)
  float* HB = SIMR;                          // 2048 x 4096
  float* GB = SIMR + 8L * 1024 * 1024;       // 2048 x 2048

  const long RN = 2048L * 512;

  // --- Fourier features + spatial bias ---
  fourier_kernel<<<8, 128, 0, stream>>>(coords, B_q,  FENCQ);
  fourier_kernel<<<8, 128, 0, stream>>>(coords, B_l0, FENC0);
  fourier_kernel<<<8, 128, 0, stream>>>(coords, B_l1, FENC1);
  rel_kernel<<<(1024 * 1024 + 255) / 256, 256, 0, stream>>>(coords, REL);

  // --- query branch: XQ = relu(fourier @ ql_w + ql_b), broadcast over batch ---
  launch_gemm(stream, FENCQ, ql_w, XQ, 1024, 512, 64, 64, 512, 512,
              0, 1.f, ql_b, 1, nullptr, nullptr, 0,
              2, 1, 0, 0, 0, 0, 1024L * 512, 0);

  // --- cross attention (mod_vec) ---
  ln_kernel<<<2048, 256, 0, stream>>>(XQ, ca_ng, ca_nb, TL, 512);
  ln_kernel<<<2048, 256, 0, stream>>>(tokens, ca_cg, ca_cb, TT, 256);
  launch_gemm(stream, TL, ca_q, QB, 2048, 128, 512, 512, 128, 128,
              0, 1.f, nullptr, 0, nullptr, nullptr, 0, 1, 1, 0,0,0,0,0,0);
  launch_gemm(stream, TT, ca_kv, KVB, 2048, 256, 256, 256, 256, 256,
              0, 1.f, nullptr, 0, nullptr, nullptr, 0, 1, 1, 0,0,0,0,0,0);
  // sim = q@k^T * (1/8) + rel, per (b,h) z=4
  launch_gemm(stream, QB, KVB, SIMR, 1024, 1024, 64, 128, 256, 1024,
              1, 0.125f, nullptr, 0, nullptr, REL, 1024,
              4, 2, 1024L * 128, 64, 1024L * 256, 64, 2L * 1024 * 1024, 1024L * 1024);
  softmax_kernel<<<4096, 256, 0, stream>>>(SIMR, 1024);
  launch_gemm(stream, SIMR, KVB + 128, OB, 1024, 64, 1024, 1024, 256, 128,
              0, 1.f, nullptr, 0, nullptr, nullptr, 0,
              4, 2, 2L * 1024 * 1024, 1024L * 1024, 1024L * 256, 64, 1024L * 128, 64);
  launch_gemm(stream, OB, ca_ow, MODV, 2048, 512, 128, 128, 512, 512,
              0, 1.f, ca_ob, 0, nullptr, nullptr, 0, 1, 1, 0,0,0,0,0,0);

  // --- transformer block (pre-LN attn + GEGLU FF, residuals), in-place on x ---
  auto run_block = [&](const Blk& bk, float* x) {
    ln_kernel<<<2048, 256, 0, stream>>>(x, bk.ng, bk.nb, TL, 512);
    launch_gemm(stream, TL, bk.tq, QB, 2048, 512, 512, 512, 512, 512,
                0, 1.f, nullptr, 0, nullptr, nullptr, 0, 1, 1, 0,0,0,0,0,0);
    launch_gemm(stream, TL, bk.tkv, KVB, 2048, 1024, 512, 512, 1024, 1024,
                0, 1.f, nullptr, 0, nullptr, nullptr, 0, 1, 1, 0,0,0,0,0,0);
    for (int b = 0; b < 2; ++b) {
      launch_gemm(stream, QB + b * 1024L * 512, KVB + b * 1024L * 1024, SIMR,
                  1024, 1024, 64, 512, 1024, 1024,
                  1, 0.125f, nullptr, 0, nullptr, nullptr, 0,
                  8, 8, 0, 64, 0, 64, 0, 1024L * 1024);
      softmax_kernel<<<8192, 256, 0, stream>>>(SIMR, 1024);
      launch_gemm(stream, SIMR, KVB + b * 1024L * 1024 + 512, OB + b * 1024L * 512,
                  1024, 64, 1024, 1024, 1024, 512,
                  0, 1.f, nullptr, 0, nullptr, nullptr, 0,
                  8, 8, 0, 1024L * 1024, 0, 64, 0, 64);
    }
    launch_gemm(stream, OB, bk.tow, x, 2048, 512, 512, 512, 512, 512,
                0, 1.f, bk.tob, 0, x, nullptr, 0, 1, 1, 0,0,0,0,0,0);
    ln_kernel<<<2048, 256, 0, stream>>>(x, bk.fng, bk.fnb, TL, 512);
    launch_gemm(stream, TL, bk.w1, HB, 2048, 4096, 512, 512, 4096, 4096,
                0, 1.f, bk.b1, 0, nullptr, nullptr, 0, 1, 1, 0,0,0,0,0,0);
    geglu_kernel<<<(int)((2048L * 2048 + 255) / 256), 256, 0, stream>>>(HB, GB, 2048L * 2048);
    launch_gemm(stream, GB, bk.w2, x, 2048, 512, 2048, 2048, 512, 512,
                0, 1.f, bk.b2, 0, x, nullptr, 0, 1, 1, 0,0,0,0,0,0);
  };

  auto run_seq_bcast = [&](const Seq& s, const float* fenc, float* x) {
    launch_gemm(stream, fenc, s.lw, x, 1024, 512, 64, 64, 512, 512,
                0, 1.f, s.lb, 1, nullptr, nullptr, 0,
                2, 1, 0, 0, 0, 0, 1024L * 512, 0);
    run_block(s.blk[0], x);
    run_block(s.blk[1], x);
  };
  auto run_seq_full = [&](const Seq& s, const float* in, float* x) {
    launch_gemm(stream, in, s.lw, x, 2048, 512, 512, 512, 512, 512,
                0, 1.f, s.lb, 1, nullptr, nullptr, 0, 1, 1, 0,0,0,0,0,0);
    run_block(s.blk[0], x);
    run_block(s.blk[1], x);
  };

  // --- layer 0 ---
  run_seq_bcast(bw0, FENC0, XB);
  run_seq_full(md0, MODV, XM);
  add_kernel<<<(int)((RN + 255) / 256), 256, 0, stream>>>(XB, XM, MODS0, RN, 1);
  // --- layer 1 ---
  run_seq_bcast(bw1, FENC1, XB);
  run_seq_full(md1, MODV, XM);
  add_kernel<<<(int)((RN + 255) / 256), 256, 0, stream>>>(XB, XM, MODS1, RN, 1);

  // --- h_v block: x = mods[1] + h_v[0]; then transformer block ---
  add_kernel<<<(int)((RN + 255) / 256), 256, 0, stream>>>(MODS1, MODS0, XV, RN, 0);
  run_block(hv, XV);

  // --- outputs: out = sum_i (h_v[i] + fenc[i]@fskip_i) @ out_i ---
  launch_gemm(stream, FENC0, fsw0, SKIP, 1024, 512, 64, 64, 512, 512,
              0, 1.f, fsb0, 0, nullptr, nullptr, 0,
              2, 1, 0, 0, 0, 0, 1024L * 512, 0);
  add_kernel<<<(int)((RN + 255) / 256), 256, 0, stream>>>(MODS0, SKIP, SKIP, RN, 0);
  outproj_kernel<<<(2048 * 3 + 255) / 256, 256, 0, stream>>>(SKIP, ow0, ob0, out, 0);

  launch_gemm(stream, FENC1, fsw1, SKIP, 1024, 512, 64, 64, 512, 512,
              0, 1.f, fsb1, 0, nullptr, nullptr, 0,
              2, 1, 0, 0, 0, 0, 1024L * 512, 0);
  add_kernel<<<(int)((RN + 255) / 256), 256, 0, stream>>>(XV, SKIP, SKIP, RN, 0);
  outproj_kernel<<<(2048 * 3 + 255) / 256, 256, 0, stream>>>(SKIP, ow1, ob1, out, 1);
}
